// SpikingLayer_50809463112001
// MI455X (gfx1250) — matrix operand order repaired and checked by reading, not compiled, not run on hardware
//
#include <hip/hip_runtime.h>

typedef __attribute__((ext_vector_type(16))) _Float16 v16h;
typedef __attribute__((ext_vector_type(8)))  _Float16 v8h;
typedef __attribute__((ext_vector_type(8)))  float    v8f;
typedef __attribute__((ext_vector_type(4)))  float    v4f;

#define T_LEN   1024
#define TAPS    100
#define WIN     128          // input-time window per 16-output tile (100 taps + 15, padded to 128)
#define WSTRIDE 136          // halfs per LDS window row (128 + 8 pad -> 272B stride, kills bank conflicts)
#define TSTRIDE 17           // floats per LDS transpose row (16 + 1 pad)

#define SEG_LEN     256      // owned time steps per block (4 segments -> 4096 waves for occupancy)
#define WARM_TILES  12       // 192 scan warm-up steps: r attenuated by e^-19.2 (< f32 eps) at segment start
#define STAGE_TILES 7        // 112 extra staged input steps so the conv window is exact at warm-up start

// One wave (32 lanes) owns 32 neuron rows for one 256-step segment.
// Per 16-step time tile:
//   vmem(32x16) = X_window(32x128, f16 in LDS ring) x Toeplitz W(128x16, f16 regs)
//   via 2 M-halves x 4 K-chunks of v_wmma_f32_16x16x32_f16, then LDS transpose
//   so each lane scans its own row's 16 consecutive time steps in registers.
__global__ __launch_bounds__(32)
void snn_wmma_kernel(const float* __restrict__ x,
                     const float* __restrict__ epsp,
                     const float* __restrict__ refk,
                     float*       __restrict__ out)
{
    __shared__ _Float16 win[32 * WSTRIDE];   // circular f16 input window, 32 rows
    __shared__ float    tr[32 * TSTRIDE];    // vmem tile transpose buffer

    const int lane     = threadIdx.x;        // 0..31 (wave32)
    const int base_row = blockIdx.x * 32;

    const int ti_store0 = (blockIdx.y * SEG_LEN) >> 4;          // first tile this block stores
    const int ti_conv0  = ti_store0 - WARM_TILES;               // first tile convolved/scanned
    int ti_begin        = ti_conv0 - STAGE_TILES;               // first tile staged into window
    if (ti_begin < 0) ti_begin = 0;                             // segment 0: times < 0 are zero
    const int ti_end    = ti_store0 + (SEG_LEN >> 4);

    // ---- Build constant Toeplitz B fragments: B[k,n] = tap(112+n-k), tap(j)=epsp[j] for 0<=j<100.
    // B fill uses contiguous k order: VGPR v, lanes 0-15: K={2v,2v+1}, N=lane; lanes 16-31: K={2v+16,2v+17}, N=lane-16.
    const int nB  = lane & 15;
    const int khi = (lane >> 4) << 4;        // +16 K-offset for upper half-wave
    v16h Bf[4];
#pragma unroll
    for (int c = 0; c < 4; ++c) {
#pragma unroll
        for (int e = 0; e < 16; ++e) {
            const int k = c * 32 + khi + e;          // K index within the 128 window
            const int j = 112 + nB - k;              // tap index
            const float w = (j >= 0 && j < TAPS) ? epsp[j] : 0.0f;
            Bf[c][e] = (_Float16)w;
        }
    }
    const float alpha = refk[1] / refk[0];           // exp(-dt/tau_mem)

    // zero the window: segment 0 reads negative-time slots as 0
    for (int i = lane; i < 32 * WSTRIDE; i += 32) win[i] = (_Float16)0.0f;
    __syncthreads();

    const float* xrow = x   + (size_t)(base_row + lane) * T_LEN;
    float*       orow = out + (size_t)(base_row + lane) * T_LEN;

    float r = 0.0f;                                   // refractory state, one row per lane

    for (int ti = ti_begin; ti < ti_end; ++ti) {
        const int t0 = ti * 16;

        // ---- stage 16 new samples of this lane's row into the circular f16 window
        {
            const v4f x0 = *(const v4f*)(xrow + t0 + 0);
            const v4f x1 = *(const v4f*)(xrow + t0 + 4);
            const v4f x2 = *(const v4f*)(xrow + t0 + 8);
            const v4f x3 = *(const v4f*)(xrow + t0 + 12);
            _Float16* dst = &win[lane * WSTRIDE + (t0 & (WIN - 1))];
            dst[ 0] = (_Float16)x0.x; dst[ 1] = (_Float16)x0.y;
            dst[ 2] = (_Float16)x0.z; dst[ 3] = (_Float16)x0.w;
            dst[ 4] = (_Float16)x1.x; dst[ 5] = (_Float16)x1.y;
            dst[ 6] = (_Float16)x1.z; dst[ 7] = (_Float16)x1.w;
            dst[ 8] = (_Float16)x2.x; dst[ 9] = (_Float16)x2.y;
            dst[10] = (_Float16)x2.z; dst[11] = (_Float16)x2.w;
            dst[12] = (_Float16)x3.x; dst[13] = (_Float16)x3.y;
            dst[14] = (_Float16)x3.z; dst[15] = (_Float16)x3.w;
        }
        __syncthreads();

        if (ti < ti_conv0) continue;                 // stage-only warm-up of the sliding window

        // ---- conv tile: window time t = t0-112+k lives at slot (t & 127) = (t0+16+k) & 127.
        // A fill uses contiguous k order: lanes 0-15 K-offs {0..15}, lanes 16-31 {16..31} (matches B fill).
        v8f acc0 = {};
        v8f acc1 = {};
        const int kadd  = (lane >> 4) << 4;          // +16 K-offset for upper half-wave
        const int arow0 = lane & 15;                 // A row this lane supplies (M-tile 0)
#pragma unroll
        for (int c = 0; c < 4; ++c) {
            const int s0 = (t0 + 16 + c * 32 + kadd) & (WIN - 1);   // 8-aligned, no mid-group wrap
            const int s1 = (s0 + 8) & (WIN - 1);
            const v8h g0a = *(const v8h*)&win[ arow0       * WSTRIDE + s0];
            const v8h g1a = *(const v8h*)&win[ arow0       * WSTRIDE + s1];
            const v8h g0b = *(const v8h*)&win[(arow0 + 16) * WSTRIDE + s0];
            const v8h g1b = *(const v8h*)&win[(arow0 + 16) * WSTRIDE + s1];
            v16h A0, A1;
#pragma unroll
            for (int e = 0; e < 8; ++e) {
                A0[e] = g0a[e]; A0[8 + e] = g1a[e];
                A1[e] = g0b[e]; A1[8 + e] = g1b[e];
            }
            acc0 = __builtin_amdgcn_wmma_f32_16x16x32_f16(false, A0, false, Bf[c],
                                                          (short)0, acc0, false, false);
            acc1 = __builtin_amdgcn_wmma_f32_16x16x32_f16(false, A1, false, Bf[c],
                                                          (short)0, acc1, false, false);
        }

        // ---- transpose vmem tile: C layout is VGPR i = row (i, i+8), lane = time col
        {
            const int rbase = (lane >> 4) << 3;      // rows 0-7 vs 8-15 within the M-tile
            const int col   = lane & 15;
#pragma unroll
            for (int i = 0; i < 8; ++i) {
                tr[(     rbase + i) * TSTRIDE + col] = acc0[i];
                tr[(16 + rbase + i) * TSTRIDE + col] = acc1[i];
            }
        }
        __syncthreads();

        // ---- refractory scan: lane owns one row, 16 sequential steps in registers
        {
            const float* trow = &tr[lane * TSTRIDE];
            float o[16];
#pragma unroll
            for (int j = 0; j < 16; ++j) {
                const float v   = trow[j] - r;
                const float nsp = fmaxf(floorf(v), 0.0f);   // THRESHOLD = 1
                r = (r + nsp) * alpha;
                o[j] = nsp;
            }
            if (ti >= ti_store0) {                   // warm-up tiles update r but store nothing
                v4f* dst = (v4f*)(orow + t0);
                const v4f o0 = { o[ 0], o[ 1], o[ 2], o[ 3] };
                const v4f o1 = { o[ 4], o[ 5], o[ 6], o[ 7] };
                const v4f o2 = { o[ 8], o[ 9], o[10], o[11] };
                const v4f o3 = { o[12], o[13], o[14], o[15] };
                __builtin_nontemporal_store(o0, dst + 0);   // keep 134MB of output out of L2
                __builtin_nontemporal_store(o1, dst + 1);   // so the input stays resident
                __builtin_nontemporal_store(o2, dst + 2);
                __builtin_nontemporal_store(o3, dst + 3);
            }
        }
    }
}

extern "C" void kernel_launch(void* const* d_in, const int* in_sizes, int n_in,
                              void* d_out, int out_size, void* d_ws, size_t ws_size,
                              hipStream_t stream) {
    const float* x    = (const float*)d_in[0];   // binary_input (1,32,1024,1024) f32
    const float* epsp = (const float*)d_in[1];   // epsp_kernel (1,100) f32
    const float* refk = (const float*)d_in[2];   // ref_kernel (100,) f32
    float* out = (float*)d_out;                  // spikes (32,1024,1024) f32

    const int rows   = out_size / T_LEN;         // 32768 neuron rows
    const int rblks  = rows / 32;                // one wave32 per block, 32 rows each
    const int segs   = T_LEN / SEG_LEN;          // 4 time segments -> 4x the waves in flight
    hipLaunchKernelGGL(snn_wmma_kernel, dim3(rblks, segs), dim3(32), 0, stream,
                       x, epsp, refk, out);
    (void)in_sizes; (void)n_in; (void)d_ws; (void)ws_size;
}
